// DGTLayer_17377437679644
// MI455X (gfx1250) — compile-verified, matrix operand order checked by reading, not run on hardware
//
#include <hip/hip_runtime.h>
#include <hip/hip_bf16.h>

typedef __attribute__((ext_vector_type(16))) _Float16 v16h;
typedef __attribute__((ext_vector_type(8)))  float    v8f;
typedef unsigned int u32;
typedef __attribute__((ext_vector_type(4))) u32 u32x4;
typedef __attribute__((ext_vector_type(8))) int i32x8;
typedef __attribute__((ext_vector_type(4))) int i32x4;

#define WAVES_PER_BLOCK 8
#define CH 16   // attention j-chunk rows staged in LDS (8 KB per tensor per buffer)

#if __has_builtin(__builtin_amdgcn_tensor_load_to_lds) && \
    __has_builtin(__builtin_amdgcn_s_wait_tensorcnt)
#define USE_TDM 1
#else
#define USE_TDM 0
#endif

#if USE_TDM
// ---------------------------------------------------------------------------
// 1-D contiguous TDM copy: global -> LDS, nElem f32 elements.
// D# built per CDNA5 ISA (CDNA5 ISA):
//  g0: count=1 | lds_addr | global_addr[56:0] | type=2
//  g1: data_size=2 (4B), tensor_dim0 = tile_dim0 = nElem, everything else 0.
// 6-arg toolchain variant: (u32x4, i32x8, i32x4, i32x4, i32x8, i32 cpol).
// Issued once per wave (wave-uniform args); tracked with TENSORcnt.
// ---------------------------------------------------------------------------
__device__ __forceinline__ void tdm_load_1d(const void* gsrc, u32 ldsByteOff,
                                            u32 nElem) {
  const unsigned long long ga = (unsigned long long)gsrc;
  u32x4 g0;
  g0[0] = 1u;                                            // count=1
  g0[1] = ldsByteOff;                                    // lds_addr (bytes)
  g0[2] = (u32)(ga & 0xFFFFFFFFu);                       // global_addr[31:0]
  g0[3] = (u32)((ga >> 32) & 0x01FFFFFFu) | (2u << 30);  // ga[56:32] | type=2
  i32x8 g1;
  g1[0] = (int)(2u << 16);                               // data_size=4B, mask=0
  g1[1] = (int)((nElem & 0xFFFFu) << 16);                // tensor_dim0[15:0]
  g1[2] = (int)((nElem >> 16) & 0xFFFFu);                // tensor_dim0[31:16]
  g1[3] = (int)((nElem & 0xFFFFu) << 16);                // tile_dim0
  g1[4] = 0; g1[5] = 0; g1[6] = 0; g1[7] = 0;
  const i32x4 z4 = {0, 0, 0, 0};
  const i32x8 z8 = {0, 0, 0, 0, 0, 0, 0, 0};
  __builtin_amdgcn_tensor_load_to_lds(g0, g1, z4, z4, z8, 0);
}
#endif

// ---------------------------------------------------------------------------
// WMMA fragment loaders (f32 memory -> f16 fragments), CDNA5 16-bit layouts.
// ---------------------------------------------------------------------------
__device__ __forceinline__ v16h load_frag_a(const float* __restrict__ A, int lda,
                                            int row0, int k0, int lane) {
  const int m    = row0 + (lane & 15);
  const int khal = (lane >> 4) << 3;                 // 0 or 8
  const float* __restrict__ ap = A + (size_t)m * lda + k0 + khal;
  v16h f;
#pragma unroll
  for (int v = 0; v < 8; ++v) {
    const int kb = ((v & 4) << 2) + ((v & 3) << 1);  // 0,2,4,6,16,18,20,22
    const float2 p = *(const float2*)(ap + kb);
    f[2 * v]     = (_Float16)p.x;
    f[2 * v + 1] = (_Float16)p.y;
  }
  return f;
}

__device__ __forceinline__ v16h load_frag_b(const float* __restrict__ B, int ldb,
                                            int k0, int col0, int lane) {
  const int n    = col0 + (lane & 15);
  const int khal = (lane >> 4) << 3;
  const float* __restrict__ bp = B + (size_t)(k0 + khal) * ldb + n;
  v16h f;
#pragma unroll
  for (int v = 0; v < 8; ++v) {
    const int kb = ((v & 4) << 2) + ((v & 3) << 1);
    f[2 * v]     = (_Float16)bp[(size_t)kb * ldb];
    f[2 * v + 1] = (_Float16)bp[(size_t)(kb + 1) * ldb];
  }
  return f;
}

// ---------------------------------------------------------------------------
// WMMA GEMM: C = relu?( alpha*(A@B) + bias + resid ); one 16x16 tile per wave.
// Epilogue specialization via template ints -> straight-line codegen.
// ---------------------------------------------------------------------------
template <int BIAS, int RES, int RELU>
__global__ void wmma_gemm(const float* __restrict__ A, const float* __restrict__ B,
                          const float* __restrict__ bias, const float* __restrict__ resid,
                          float* __restrict__ C, int M, int N, int K, float alpha) {
  const int lane   = threadIdx.x & 31;
  const int wave   = threadIdx.x >> 5;
  const int tilesN = N >> 4;
  const int nTiles = (M >> 4) * tilesN;
  const int tile   = blockIdx.x * WAVES_PER_BLOCK + wave;
  if (tile >= nTiles) return;                        // wave-uniform
  const int tm = (tile / tilesN) << 4;
  const int tn = (tile % tilesN) << 4;

  v8f acc = {};
  for (int k0 = 0; k0 < K; k0 += 32) {
    const v16h a = load_frag_a(A, K, tm, k0, lane);
    const v16h b = load_frag_b(B, N, k0, tn, lane);
    acc = __builtin_amdgcn_wmma_f32_16x16x32_f16(false, a, false, b,
                                                 (short)0, acc, false, false);
  }

  const int col  = tn + (lane & 15);
  const int row0 = tm + ((lane >> 4) << 3);
  const float bcol = BIAS ? bias[col] : 0.0f;
  float* __restrict__ cp = C + (size_t)row0 * N + col;
  const float* __restrict__ rp = RES ? (resid + (size_t)row0 * N + col) : nullptr;
#pragma unroll
  for (int r = 0; r < 8; ++r) {
    float v = acc[r] * alpha + bcol;
    if (RES)  v += rp[(size_t)r * N];
    if (RELU) v = fmaxf(v, 0.0f);
    cp[(size_t)r * N] = v;
  }
}

// ---------------------------------------------------------------------------
// Fused biased attention, online softmax over j per channel d.
// One block per (b,i); 128 threads. e_att/e_val chunks (contiguous 8 KB)
// double-buffered into LDS via TDM (tensor_load_to_lds + s_wait_tensorcnt);
// wave0 issues/waits, barriers publish. K/V re-reads are L2-resident.
// ---------------------------------------------------------------------------
__global__ void attn_fused(const float* __restrict__ Q, const float* __restrict__ Kb,
                           const float* __restrict__ Vb,
                           const float* __restrict__ e_att, const float* __restrict__ e_val,
                           float* __restrict__ out, int n) {
  extern __shared__ float smem[];
  float* Qs  = smem;                   // 128
  float* qk  = smem + 128;             // up to 256*8
  float* eaS = smem + 128 + 2048;      // 2 * CH*128
  float* evS = eaS + 2 * CH * 128;     // 2 * CH*128
  const u32 EA_OFF = (128 + 2048) * 4;           // byte offsets for TDM lds_addr
  const u32 EV_OFF = EA_OFF + 2 * CH * 128 * 4;

  const int tid = threadIdx.x;                   // 0..127
  const bool w0 = (tid >> 5) == 0;               // wave 0 (full wave)
  const int b   = blockIdx.x / n;
  const size_t rowQ = (size_t)blockIdx.x * 128;
  const float* ea = e_att + (size_t)blockIdx.x * n * 128;
  const float* ev = e_val + (size_t)blockIdx.x * n * 128;
  const int nc = n / CH;

#if USE_TDM
  if (w0) {                                      // prefill chunk 0 -> buffer 0
    tdm_load_1d(ea, EA_OFF, CH * 128);
    tdm_load_1d(ev, EV_OFF, CH * 128);
  }
#endif

  Qs[tid] = Q[rowQ + tid];
  __syncthreads();

  // per-head QK dots (K pre-scaled by dk^-0.5); overlaps the first TDM fill
  const int nh = n * 8;
  for (int e = tid; e < nh; e += 128) {
    const int j = e >> 3, h = e & 7;
    const float4* kp = (const float4*)(Kb + ((size_t)(b * n + j)) * 128 + h * 16);
    const float4* qp = (const float4*)(Qs + h * 16);
    float s = 0.0f;
#pragma unroll
    for (int q = 0; q < 4; ++q) {
      const float4 kv = kp[q], qv = qp[q];
      s += kv.x * qv.x + kv.y * qv.y + kv.z * qv.z + kv.w * qv.w;
    }
    qk[e] = s;
  }
  __syncthreads();

  const int d = tid;
  const int h = d >> 4;
  const float* vp = Vb + (size_t)b * n * 128 + d;
  float m = -3.0e38f, Z = 0.0f, acc = 0.0f;

  for (int c = 0; c < nc; ++c) {
    const int buf = c & 1;
#if USE_TDM
    if (w0) {
      if (c + 1 < nc) {                          // issue next chunk into other buffer
        tdm_load_1d(ea + (size_t)(c + 1) * CH * 128,
                    EA_OFF + (u32)(buf ^ 1) * CH * 128 * 4, CH * 128);
        tdm_load_1d(ev + (size_t)(c + 1) * CH * 128,
                    EV_OFF + (u32)(buf ^ 1) * CH * 128 * 4, CH * 128);
        __builtin_amdgcn_s_wait_tensorcnt(2);    // chunk c landed; c+1 in flight
      } else {
        __builtin_amdgcn_s_wait_tensorcnt(0);
      }
    }
#else
    for (int e = tid; e < CH * 128; e += 128) {  // synchronous fallback staging
      eaS[buf * CH * 128 + e] = ea[(size_t)c * CH * 128 + e];
      evS[buf * CH * 128 + e] = ev[(size_t)c * CH * 128 + e];
    }
#endif
    __syncthreads();                             // publish buffer to all waves

    const float* eab = eaS + buf * CH * 128 + d;
    const float* evb = evS + buf * CH * 128 + d;
    const int j0 = c * CH;
#pragma unroll
    for (int jj = 0; jj < CH; ++jj) {
      const float s = qk[(j0 + jj) * 8 + h] + eab[jj * 128];
      if (s > m) {                               // rare after warmup
        const float sc = __expf(m - s);
        Z *= sc; acc *= sc; m = s;
      }
      const float p = __expf(s - m);
      const float w = vp[(size_t)(j0 + jj) * 128] + evb[jj * 128];
      Z   += p;
      acc += p * w;
    }
    __syncthreads();                             // done reading before refill
  }
  out[rowQ + d] = acc / Z;
}

// ---------------------------------------------------------------------------
// BatchNorm (training): per-column mean/rstd over R rows, then apply.
// ---------------------------------------------------------------------------
__global__ void bn_stats(const float* __restrict__ X, int R,
                         float* __restrict__ mean, float* __restrict__ rstd) {
  __shared__ float s1[256], s2[256];
  const int c = blockIdx.x;
  float a = 0.0f, q = 0.0f;
  for (int r = threadIdx.x; r < R; r += 256) {
    const float v = X[(size_t)r * 128 + c];
    a += v; q += v * v;
  }
  s1[threadIdx.x] = a; s2[threadIdx.x] = q;
  __syncthreads();
  for (int off = 128; off > 0; off >>= 1) {
    if (threadIdx.x < off) {
      s1[threadIdx.x] += s1[threadIdx.x + off];
      s2[threadIdx.x] += s2[threadIdx.x + off];
    }
    __syncthreads();
  }
  if (threadIdx.x == 0) {
    const float mu  = s1[0] / (float)R;
    const float var = s2[0] / (float)R - mu * mu;
    mean[c] = mu;
    rstd[c] = rsqrtf(var + 1e-5f);
  }
}

__global__ void bn_apply(const float* __restrict__ X, const float* __restrict__ mean,
                         const float* __restrict__ rstd, const float* __restrict__ g,
                         const float* __restrict__ be, float* __restrict__ Y, int total) {
  const int idx = blockIdx.x * blockDim.x + threadIdx.x;
  if (idx >= total) return;
  const int c = idx & 127;
  Y[idx] = (X[idx] - mean[c]) * rstd[c] * g[c] + be[c];
}

// ---------------------------------------------------------------------------
// Launcher
// ---------------------------------------------------------------------------
extern "C" void kernel_launch(void* const* d_in, const int* in_sizes, int n_in,
                              void* d_out, int out_size, void* d_ws, size_t ws_size,
                              hipStream_t stream) {
  (void)in_sizes; (void)n_in; (void)out_size; (void)ws_size;
  const int Bb = 8, Nn = 256, Mm = 128;

  float* ws = (float*)d_ws;
  const size_t RMAX = 2048;
  float* Qb   = ws; ws += RMAX * 128;
  float* Kbuf = ws; ws += RMAX * 128;
  float* Vbuf = ws; ws += RMAX * 128;
  float* attn = ws; ws += RMAX * 128;
  float* r1   = ws; ws += RMAX * 128;
  float* h1   = ws; ws += RMAX * 128;
  float* f1   = ws; ws += RMAX * 256;
  float* mean = ws; ws += 128;
  float* rstd = ws; ws += 128;

  const int SMEM_BYTES = (128 + 2048 + 4 * CH * 128) * 4;   // 41,472 B

  auto grid_of = [](int M, int N) {
    const int tiles = (M >> 4) * (N >> 4);
    return (tiles + WAVES_PER_BLOCK - 1) / WAVES_PER_BLOCK;
  };

  for (int blk = 0; blk < 2; ++blk) {
    const int n = blk ? Mm : Nn;
    const int R = Bb * n;
    const float* h  = (const float*)d_in[blk];
    const float* ea = (const float*)d_in[2 + 2 * blk];
    const float* ev = (const float*)d_in[3 + 2 * blk];
    const int p = 6 + 13 * blk;
    const float* Wq  = (const float*)d_in[p + 0];
    const float* Wk  = (const float*)d_in[p + 1];
    const float* Wv  = (const float*)d_in[p + 2];
    const float* Wl  = (const float*)d_in[p + 3];
    const float* bl  = (const float*)d_in[p + 4];
    const float* g1  = (const float*)d_in[p + 5];
    const float* be1 = (const float*)d_in[p + 6];
    const float* Wf1 = (const float*)d_in[p + 7];
    const float* bf1 = (const float*)d_in[p + 8];
    const float* Wf2 = (const float*)d_in[p + 9];
    const float* bf2 = (const float*)d_in[p + 10];
    const float* g2  = (const float*)d_in[p + 11];
    const float* be2 = (const float*)d_in[p + 12];
    float* outp = (float*)d_out + (blk ? (size_t)Bb * Nn * 128 : 0);

    // Projections (dk = 16 -> K pre-scaled by dk^-0.5 = 0.25)
    wmma_gemm<0,0,0><<<grid_of(R,128), 256, 0, stream>>>(h, Wq, nullptr, nullptr, Qb,   R, 128, 128, 1.0f);
    wmma_gemm<0,0,0><<<grid_of(R,128), 256, 0, stream>>>(h, Wk, nullptr, nullptr, Kbuf, R, 128, 128, 0.25f);
    wmma_gemm<0,0,0><<<grid_of(R,128), 256, 0, stream>>>(h, Wv, nullptr, nullptr, Vbuf, R, 128, 128, 1.0f);

    // Fused attention (TDM double-buffered streaming of e_att/e_val)
    attn_fused<<<R, 128, SMEM_BYTES, stream>>>(Qb, Kbuf, Vbuf, ea, ev, attn, n);

    // Output projection + residual, BN1
    wmma_gemm<1,1,0><<<grid_of(R,128), 256, 0, stream>>>(attn, Wl, bl, h, r1, R, 128, 128, 1.0f);
    bn_stats<<<128, 256, 0, stream>>>(r1, R, mean, rstd);
    bn_apply<<<(R * 128) / 256, 256, 0, stream>>>(r1, mean, rstd, g1, be1, h1, R * 128);

    // FFN + residual, BN2 -> output
    wmma_gemm<1,0,1><<<grid_of(R,256), 256, 0, stream>>>(h1, Wf1, bf1, nullptr, f1, R, 256, 128, 1.0f);
    wmma_gemm<1,1,0><<<grid_of(R,128), 256, 0, stream>>>(f1, Wf2, bf2, h1, r1, R, 128, 256, 1.0f);
    bn_stats<<<128, 256, 0, stream>>>(r1, R, mean, rstd);
    bn_apply<<<(R * 128) / 256, 256, 0, stream>>>(r1, mean, rstd, g2, be2, outp, R * 128);
  }
}
